// SimplifiedMambaBlock_51264729645604
// MI455X (gfx1250) — matrix operand order repaired and checked by reading, compiled, not run on hardware
//
#include <hip/hip_runtime.h>
#include <hip/hip_bf16.h>

// ---------------------------------------------------------------------------
// SimplifiedMambaBlock for MI455X (gfx1250, wave32, WMMA bf16 path)
// D_MODEL=1024, D_STATE=16, D_INNER=2048, B=2, S=4096  -> M = B*S = 8192 rows
// ---------------------------------------------------------------------------

typedef __attribute__((ext_vector_type(16))) __bf16 v16bf;
typedef __attribute__((ext_vector_type(8)))  __bf16 v8bf;
typedef __attribute__((ext_vector_type(4)))  __bf16 v4bf;
typedef __attribute__((ext_vector_type(8)))  float  v8f;

#define D_MODEL 1024
#define D_STATE 16
#define D_INNER 2048
#define BATCH   2
#define SEQ     4096
#define MROWS   (BATCH * SEQ)        // 8192
#define EPS     1e-5f
#define NCHUNK  64                    // scan chunks
#define CHLEN   (SEQ / NCHUNK)        // 64

// -------------------------- LayerNorm -> bf16 ------------------------------
__global__ __launch_bounds__(256)
void layernorm_bf16_kernel(const float* __restrict__ x,
                           const float* __restrict__ w,
                           const float* __restrict__ b,
                           __bf16* __restrict__ xn) {
    const int row = blockIdx.x;
    const int tid = threadIdx.x;
    const float4* xr = (const float4*)(x + (size_t)row * D_MODEL);
    float4 v = xr[tid];                       // 256 threads * 4 = 1024
    float s  = v.x + v.y + v.z + v.w;
    float s2 = v.x*v.x + v.y*v.y + v.z*v.z + v.w*v.w;
    #pragma unroll
    for (int off = 16; off > 0; off >>= 1) {
        s  += __shfl_xor(s,  off, 32);
        s2 += __shfl_xor(s2, off, 32);
    }
    __shared__ float ps[8], ps2[8];
    const int wv = tid >> 5, ln = tid & 31;
    if (ln == 0) { ps[wv] = s; ps2[wv] = s2; }
    __syncthreads();
    if (tid == 0) {
        float t = 0.f, t2 = 0.f;
        #pragma unroll
        for (int i = 0; i < 8; ++i) { t += ps[i]; t2 += ps2[i]; }
        ps[0] = t; ps2[0] = t2;
    }
    __syncthreads();
    const float mean = ps[0] * (1.0f / D_MODEL);
    const float var  = ps2[0] * (1.0f / D_MODEL) - mean * mean;
    const float inv  = rsqrtf(var + EPS);
    const float4 wv4 = ((const float4*)w)[tid];
    const float4 bv4 = ((const float4*)b)[tid];
    v4bf o;
    o[0] = (__bf16)((v.x - mean) * inv * wv4.x + bv4.x);
    o[1] = (__bf16)((v.y - mean) * inv * wv4.y + bv4.y);
    o[2] = (__bf16)((v.z - mean) * inv * wv4.z + bv4.z);
    o[3] = (__bf16)((v.w - mean) * inv * wv4.w + bv4.w);
    ((v4bf*)(xn + (size_t)row * D_MODEL))[tid] = o;
}

// ----------------------- f32 -> bf16 conversion ----------------------------
__global__ __launch_bounds__(256)
void cvt_bf16_kernel(const float* __restrict__ in, __bf16* __restrict__ out, int n) {
    int i = blockIdx.x * 256 + threadIdx.x;
    if (i < n) out[i] = (__bf16)in[i];
}

// ------------------- WMMA bf16 GEMM: out = A * Bw^T + bias (+resid) --------
// A  : [M, K] bf16 row-major (activations)
// Bw : [N, K] bf16 row-major (weight; K contiguous -> natural B-operand)
// Block = 256 threads = 8 waves. Block tile 128(M) x 64(N); each wave 16x64.
// Ping-pong software pipeline over two named fragment sets: straight-line
// steady state, no conditional buffer swaps (no v_mov churn).

__device__ __forceinline__ v16bf load_frag16(const __bf16* p0, const __bf16* p1) {
    v8bf lo = *(const v8bf*)p0;
    v8bf hi = *(const v8bf*)p1;
    v16bf f;
    #pragma unroll
    for (int i = 0; i < 8; ++i) { f[i] = lo[i]; f[i + 8] = hi[i]; }
    return f;
}

__global__ __launch_bounds__(256, 2)
void gemm_bf16_wmma_kernel(const __bf16* __restrict__ A,
                           const __bf16* __restrict__ Bw,
                           const float* __restrict__ bias,
                           const float* __restrict__ resid,
                           void* __restrict__ outp,
                           int M, int N, int K,
                           int out_is_bf16, int add_resid) {
    const int wave = threadIdx.x >> 5;
    const int lane = threadIdx.x & 31;
    const int halo = lane >> 4;                   // 0: lanes 0-15, 1: lanes 16-31
    const int l16  = lane & 15;
    const int rowBase = blockIdx.y * 128 + wave * 16;
    const int colBase = blockIdx.x * 64;

    // A fragment 16x32 (ISA layout): lanes 0-15 hold K[k0..+7] / K[k0+16..+23],
    // lanes 16-31 hold K[k0+8..+15] / K[k0+24..+31]  -> fold 8*halo into base.
    const __bf16* arow = A + (size_t)(rowBase + l16) * K + 8 * halo;
    // B fragment 32x16: lane col = lane%16, same K split as A (8*halo / +16).
    const __bf16* bcol[4];
    #pragma unroll
    for (int t = 0; t < 4; ++t)
        bcol[t] = Bw + (size_t)(colBase + t * 16 + l16) * K + 8 * halo;

    v8f acc[4] = {};   // 4 tiles of 16x16 f32
    v16bf a0, a1, b0[4], b1[4];

    // prologue: stage 0 holds k = 0
    a0 = load_frag16(arow, arow + 16);
    #pragma unroll
    for (int t = 0; t < 4; ++t)
        b0[t] = load_frag16(bcol[t], bcol[t] + 16);

    // steady state: consume steps k and k+32, preload k+64. K/32 is even.
    int k = 0;
    #pragma unroll 1
    for (; k < K - 64; k += 64) {
        const int k1 = k + 32;
        a1 = load_frag16(arow + k1, arow + k1 + 16);
        #pragma unroll
        for (int t = 0; t < 4; ++t)
            b1[t] = load_frag16(bcol[t] + k1, bcol[t] + k1 + 16);
        __builtin_prefetch(arow + k + 128, 0, 3);

        #pragma unroll
        for (int t = 0; t < 4; ++t)
            acc[t] = __builtin_amdgcn_wmma_f32_16x16x32_bf16(
                false, a0, false, b0[t], (short)0, acc[t], false, false);

        const int k2 = k + 64;
        a0 = load_frag16(arow + k2, arow + k2 + 16);
        #pragma unroll
        for (int t = 0; t < 4; ++t)
            b0[t] = load_frag16(bcol[t] + k2, bcol[t] + k2 + 16);

        #pragma unroll
        for (int t = 0; t < 4; ++t)
            acc[t] = __builtin_amdgcn_wmma_f32_16x16x32_bf16(
                false, a1, false, b1[t], (short)0, acc[t], false, false);
    }

    // tail: a0/b0 hold step K-64; load and consume K-32.
    {
        const int k1 = K - 32;
        a1 = load_frag16(arow + k1, arow + k1 + 16);
        #pragma unroll
        for (int t = 0; t < 4; ++t)
            b1[t] = load_frag16(bcol[t] + k1, bcol[t] + k1 + 16);
        #pragma unroll
        for (int t = 0; t < 4; ++t)
            acc[t] = __builtin_amdgcn_wmma_f32_16x16x32_bf16(
                false, a0, false, b0[t], (short)0, acc[t], false, false);
        #pragma unroll
        for (int t = 0; t < 4; ++t)
            acc[t] = __builtin_amdgcn_wmma_f32_16x16x32_bf16(
                false, a1, false, b1[t], (short)0, acc[t], false, false);
    }

    // Epilogue. C/D layout: VGPR r holds row M = r + 8*(lane/16), col = lane%16.
    const int mLane = rowBase + 8 * halo;
    #pragma unroll
    for (int t = 0; t < 4; ++t) {
        const int col = colBase + t * 16 + l16;
        const float bv = bias[col];
        #pragma unroll
        for (int r = 0; r < 8; ++r) {
            const int row = mLane + r;
            float v = acc[t][r] + bv;
            const size_t idx = (size_t)row * N + col;
            if (add_resid) v += resid[idx];
            if (out_is_bf16) ((__bf16*)outp)[idx] = (__bf16)v;
            else             ((float*)outp)[idx]  = v;
        }
    }
}

// ------------- gates: a = sigmoid(h@A), u = h@B_ssm  (K=2048, N=16 each) ----
__global__ __launch_bounds__(256)
void gates_kernel(const __bf16* __restrict__ h,
                  const float* __restrict__ Amat,
                  const float* __restrict__ Bmat,
                  float* __restrict__ ag, float* __restrict__ ud) {
    __shared__ float hrow[D_INNER];
    __shared__ float part[8][32];
    const int m = blockIdx.x;
    const int tid = threadIdx.x;
    // stage h row in LDS (f32)
    v8bf hv = *(const v8bf*)(h + (size_t)m * D_INNER + tid * 8);
    #pragma unroll
    for (int i = 0; i < 8; ++i) hrow[tid * 8 + i] = (float)hv[i];
    __syncthreads();

    const int wv = tid >> 5, ln = tid & 31;
    const int col = ln & 15;
    const float* W = (ln < 16) ? Amat : Bmat;
    float acc = 0.f;
    const int e0 = wv * 256;
    for (int i = 0; i < 256; ++i) {
        const int e = e0 + i;
        acc += hrow[e] * W[e * D_STATE + col];
    }
    part[wv][ln] = acc;
    __syncthreads();
    if (tid < 32) {
        float s = 0.f;
        #pragma unroll
        for (int i = 0; i < 8; ++i) s += part[i][tid];
        if (tid < 16) ag[(size_t)m * D_STATE + tid] = 1.f / (1.f + expf(-s));
        else          ud[(size_t)m * D_STATE + (tid - 16)] = s;
    }
}

// --------------- 3-phase chunked scan of s_t = a_t*s_{t-1} + u_t ------------
// 32 chains (b,n); composition (a,u)∘(a',u') = (a*a', u' + a'*u)
__global__ void scan_chunks_kernel(const float* __restrict__ ag,
                                   const float* __restrict__ ud,
                                   float* __restrict__ chA, float* __restrict__ chU) {
    const int c = threadIdx.x;                                  // 0..31 chain
    const int j = blockIdx.x * blockDim.y + threadIdx.y;        // 0..63 chunk
    const int b = c >> 4, n = c & 15;
    const size_t base = ((size_t)b * SEQ + (size_t)j * CHLEN) * D_STATE + n;
    float aP = 1.f, uA = 0.f;
    for (int t = 0; t < CHLEN; ++t) {
        const float a = ag[base + (size_t)t * D_STATE];
        const float u = ud[base + (size_t)t * D_STATE];
        aP *= a; uA = uA * a + u;
    }
    chA[j * 32 + c] = aP;
    chU[j * 32 + c] = uA;
}

__global__ void scan_tops_kernel(const float* __restrict__ chA,
                                 const float* __restrict__ chU,
                                 float* __restrict__ pre) {
    const int c = threadIdx.x;   // 32 threads
    float s = 0.f;
    for (int j = 0; j < NCHUNK; ++j) {
        pre[j * 32 + c] = s;
        s = s * chA[j * 32 + c] + chU[j * 32 + c];
    }
}

__global__ void scan_apply_kernel(const float* __restrict__ ag,
                                  const float* __restrict__ ud,
                                  const float* __restrict__ pre,
                                  float* __restrict__ st) {
    const int c = threadIdx.x;
    const int j = blockIdx.x * blockDim.y + threadIdx.y;
    const int b = c >> 4, n = c & 15;
    const size_t base = ((size_t)b * SEQ + (size_t)j * CHLEN) * D_STATE + n;
    float s = pre[j * 32 + c];
    for (int t = 0; t < CHLEN; ++t) {
        const size_t idx = base + (size_t)t * D_STATE;
        s = s * ag[idx] + ud[idx];
        st[idx] = s;
    }
}

// ----------------- readout: y[m,e] = sum_n st[m,n]*C[e,n] -> bf16 -----------
__global__ __launch_bounds__(256)
void readout_kernel(const float* __restrict__ st,
                    const float* __restrict__ Cmat,
                    __bf16* __restrict__ y) {
    __shared__ float sl[D_STATE];
    const int m = blockIdx.x;
    const int tid = threadIdx.x;
    if (tid < D_STATE) sl[tid] = st[(size_t)m * D_STATE + tid];
    __syncthreads();
    #pragma unroll
    for (int i = 0; i < 8; ++i) {
        const int e = tid + i * 256;
        const float4* cp = (const float4*)(Cmat + (size_t)e * D_STATE);
        float acc = 0.f;
        #pragma unroll
        for (int q = 0; q < 4; ++q) {
            float4 cv = cp[q];
            acc += sl[q*4+0]*cv.x + sl[q*4+1]*cv.y + sl[q*4+2]*cv.z + sl[q*4+3]*cv.w;
        }
        y[(size_t)m * D_INNER + e] = (__bf16)acc;
    }
}

// ---------------------------------------------------------------------------
extern "C" void kernel_launch(void* const* d_in, const int* in_sizes, int n_in,
                              void* d_out, int out_size, void* d_ws, size_t ws_size,
                              hipStream_t stream) {
    const float* x      = (const float*)d_in[0];
    const float* norm_w = (const float*)d_in[1];
    const float* norm_b = (const float*)d_in[2];
    const float* W_in   = (const float*)d_in[3];
    const float* b_in   = (const float*)d_in[4];
    const float* Amat   = (const float*)d_in[5];
    const float* B_ssm  = (const float*)d_in[6];
    const float* C_ssm  = (const float*)d_in[7];
    const float* W_out  = (const float*)d_in[8];
    const float* b_out  = (const float*)d_in[9];

    char* ws = (char*)d_ws;
    // workspace layout (bytes)
    __bf16* xn     = (__bf16*)(ws + 0);                         // 8192*1024*2 = 16 MiB
    __bf16* win_bf = (__bf16*)(ws + 16777216);                  // 2048*1024*2 =  4 MiB
    __bf16* wout_bf= (__bf16*)(ws + 20971520);                  // 1024*2048*2 =  4 MiB
    __bf16* h      = (__bf16*)(ws + 25165824);                  // 8192*2048*2 = 32 MiB
    float*  ag     = (float*) (ws + 58720256);                  // 8192*16*4
    float*  ud     = (float*) (ws + 59244544);
    float*  chA    = (float*) (ws + 59768832);
    float*  chU    = (float*) (ws + 59777024);
    float*  pre    = (float*) (ws + 59785216);
    float*  st     = (float*) (ws + 59793408);                  // states
    __bf16* y      = (__bf16*)(ws + 60317696);                  // 8192*2048*2 = 32 MiB
    (void)in_sizes; (void)n_in; (void)out_size; (void)ws_size;

    // 1. LayerNorm -> bf16 activations
    layernorm_bf16_kernel<<<MROWS, 256, 0, stream>>>(x, norm_w, norm_b, xn);

    // 2. weights -> bf16
    cvt_bf16_kernel<<<(D_INNER * D_MODEL + 255) / 256, 256, 0, stream>>>(W_in,  win_bf, D_INNER * D_MODEL);
    cvt_bf16_kernel<<<(D_MODEL * D_INNER + 255) / 256, 256, 0, stream>>>(W_out, wout_bf, D_MODEL * D_INNER);

    // 3. in-proj: h = xn @ W_in^T + b_in   (M=8192, N=2048, K=1024) -> bf16
    gemm_bf16_wmma_kernel<<<dim3(D_INNER / 64, MROWS / 128), 256, 0, stream>>>(
        xn, win_bf, b_in, nullptr, h, MROWS, D_INNER, D_MODEL, /*bf16*/1, /*resid*/0);

    // 4. gates a = sigmoid(h@A), drive u = h@B_ssm
    gates_kernel<<<MROWS, 256, 0, stream>>>(h, Amat, B_ssm, ag, ud);

    // 5. chunked parallel scan of the diagonal recurrence
    scan_chunks_kernel<<<NCHUNK / 8, dim3(32, 8), 0, stream>>>(ag, ud, chA, chU);
    scan_tops_kernel<<<1, 32, 0, stream>>>(chA, chU, pre);
    scan_apply_kernel<<<NCHUNK / 8, dim3(32, 8), 0, stream>>>(ag, ud, pre, st);

    // 6. readout y = states @ C^T -> bf16
    readout_kernel<<<MROWS, 256, 0, stream>>>(st, C_ssm, y);

    // 7. out-proj: out = y @ W_out^T + b_out + x   (M=8192, N=1024, K=2048) -> f32
    gemm_bf16_wmma_kernel<<<dim3(D_MODEL / 64, MROWS / 128), 256, 0, stream>>>(
        y, wout_bf, b_out, x, d_out, MROWS, D_MODEL, D_INNER, /*bf16*/0, /*resid*/1);
}
